// AttentionGRUDecoder_51135880626329
// MI455X (gfx1250) — compile-verified, matrix operand order checked by reading, not run on hardware
//
#include <hip/hip_runtime.h>
#include <hip/hip_bf16.h>

// ---------------------------------------------------------------------------
// AttentionGRUDecoder on MI455X (gfx1250): bf16 WMMA GEMMs, f32 accumulation.
// B=512, S=64, H=A=O=1024, ATT=64, T(max_length)=128 (known from setup).
//
// All weights (18 MB bf16) + encoder (128 MB f32) stay resident in the 192 MB
// L2; per-wave tiles are 32 rows wide so each weight fragment feeds 2 WMMAs,
// and the 256 MB output stream is written with non-temporal stores to avoid
// evicting the hot working set.
// ---------------------------------------------------------------------------

typedef __bf16 bf16;
typedef __attribute__((ext_vector_type(16))) __bf16 v16bf;
typedef __attribute__((ext_vector_type(8)))  __bf16 v8bf;
typedef __attribute__((ext_vector_type(8)))  float  v8f;

#define BATCH 512
#define SEQ   64
#define HID   1024
#define TSTEP 128

__device__ __forceinline__ unsigned lane_id() { return threadIdx.x & 31u; }

// Load a 16x32 bf16 fragment (A or B of an NT GEMM) from a row-major matrix.
// ISA 16-bit layout: lanes 0-15 carry rows 0..15 with K=0..7 (VGPR0-3) and
// K=16..23 (VGPR4-7); lanes 16-31 carry the same rows at K+8.
__device__ __forceinline__ v16bf load_frag(const bf16* __restrict__ tile, int ld) {
    unsigned l    = lane_id();
    unsigned m    = l & 15u;
    unsigned half = l >> 4;
    const bf16* p = tile + (size_t)m * ld + half * 8u;
    v8bf lo = *(const v8bf*)(p);
    v8bf hi = *(const v8bf*)(p + 16);
    v16bf f;
#pragma unroll
    for (int i = 0; i < 8; ++i) { f[i] = lo[i]; f[i + 8] = hi[i]; }
    return f;
}

__device__ __forceinline__ v8f wmma_bf16(v16bf a, v16bf b, v8f c) {
    // 8 args: (neg_a, A, neg_b, B, c_mod, C, reuse_a, reuse_b)
    return __builtin_amdgcn_wmma_f32_16x16x32_bf16(false, a, false, b, (short)0, c,
                                                   false, false);
}

__device__ __forceinline__ float sigmoidf_fast(float x) {
    return 1.0f / (1.0f + __expf(-x));
}

// ---------------------------------------------------------------------------
// Utility kernels
// ---------------------------------------------------------------------------
__global__ void __launch_bounds__(256) k_f32_to_bf16(const float* __restrict__ src,
                                                     bf16* __restrict__ dst, int n) {
    int i = blockIdx.x * 256 + threadIdx.x;
    if (i < n) dst[i] = (bf16)src[i];
}

__global__ void __launch_bounds__(256) k_zero_bf16(bf16* __restrict__ dst, int n) {
    int i = blockIdx.x * 256 + threadIdx.x;
    if (i < n) dst[i] = (bf16)0.0f;
}

// ---------------------------------------------------------------------------
// Step kernel 1: attn logits [B,64] = [dec|h] @ Wa^T + ba, fused row softmax.
// Block = 128 threads = 4 waves; each wave computes one 16x16 logit tile,
// block covers 16 batch rows x all 64 attention columns.
// ---------------------------------------------------------------------------
__global__ void __launch_bounds__(128)
k_attn_logits(const bf16* __restrict__ dec, const bf16* __restrict__ h,
              const bf16* __restrict__ Wa, const float* __restrict__ ba,
              float* __restrict__ attn_w) {
    __shared__ float logits[16][65];
    const int row0 = blockIdx.x * 16;
    const int n0   = (threadIdx.x >> 5) * 16;

    v8f acc = {};
    for (int k0 = 0; k0 < 1024; k0 += 32) {            // K segment 0: dec_in
        v16bf a = load_frag(dec + (size_t)row0 * 1024 + k0, 1024);
        v16bf b = load_frag(Wa + (size_t)n0 * 2048 + k0, 2048);
        acc = wmma_bf16(a, b, acc);
    }
    for (int k0 = 0; k0 < 1024; k0 += 32) {            // K segment 1: hidden
        v16bf a = load_frag(h + (size_t)row0 * 1024 + k0, 1024);
        v16bf b = load_frag(Wa + (size_t)n0 * 2048 + 1024 + k0, 2048);
        acc = wmma_bf16(a, b, acc);
    }
    unsigned l  = lane_id();
    unsigned nn = n0 + (l & 15u);
    unsigned mo = (l >> 4) * 8u;
    float bias  = ba[nn];
#pragma unroll
    for (int r = 0; r < 8; ++r) logits[mo + r][nn] = acc[r] + bias;
    __syncthreads();

    if (threadIdx.x < 16) {
        int row = threadIdx.x;
        float mx = -3.0e38f;
#pragma unroll 8
        for (int s = 0; s < 64; ++s) mx = fmaxf(mx, logits[row][s]);
        float sum = 0.0f;
#pragma unroll 8
        for (int s = 0; s < 64; ++s) sum += __expf(logits[row][s] - mx);
        float inv = 1.0f / sum;
        float* out = attn_w + (size_t)(row0 + row) * 64;
#pragma unroll 8
        for (int s = 0; s < 64; ++s) out[s] = __expf(logits[row][s] - mx) * inv;
    }
}

// ---------------------------------------------------------------------------
// Step kernel 2: attn_applied[b,d] = sum_s w[b,s] * enc[b,s,d]  (per-batch GEMV)
// ---------------------------------------------------------------------------
__global__ void __launch_bounds__(256)
k_attn_apply(const float* __restrict__ attn_w, const float* __restrict__ enc,
             bf16* __restrict__ attn_bf) {
    int b = blockIdx.x;
    int d = blockIdx.y * 256 + threadIdx.x;
    const float* w = attn_w + (size_t)b * SEQ;
    const float* e = enc + ((size_t)b * SEQ) * 1024 + d;
    float acc = 0.0f;
#pragma unroll 8
    for (int s = 0; s < SEQ; ++s) acc = fmaf(w[s], e[(size_t)s * 1024], acc);
    attn_bf[(size_t)b * 1024 + d] = (bf16)acc;
}

// ---------------------------------------------------------------------------
// Step kernel 3: generic NT GEMM, K = 1024(x0) + 1024(x1), bf16 out + opt ReLU.
// Block = 256 threads = 8 waves; each wave computes a 32x64 strip of C so every
// weight fragment feeds two WMMAs (halves B traffic from L2).
// ---------------------------------------------------------------------------
__global__ void __launch_bounds__(256)
k_gemm_2seg(const bf16* __restrict__ x0, const bf16* __restrict__ x1,
            const bf16* __restrict__ W, const float* __restrict__ bias,
            bf16* __restrict__ out, int N, int relu_flag) {
    const int row0 = blockIdx.x * 32;
    const int n0   = (blockIdx.y * 8 + (threadIdx.x >> 5)) * 64;
    v8f acc0[4] = {{}, {}, {}, {}};
    v8f acc1[4] = {{}, {}, {}, {}};

    for (int k0 = 0; k0 < 1024; k0 += 32) {
        v16bf a0 = load_frag(x0 + (size_t)row0 * 1024 + k0, 1024);
        v16bf a1 = load_frag(x0 + (size_t)(row0 + 16) * 1024 + k0, 1024);
#pragma unroll
        for (int j = 0; j < 4; ++j) {
            v16bf b = load_frag(W + (size_t)(n0 + j * 16) * 2048 + k0, 2048);
            acc0[j] = wmma_bf16(a0, b, acc0[j]);
            acc1[j] = wmma_bf16(a1, b, acc1[j]);
        }
    }
    for (int k0 = 0; k0 < 1024; k0 += 32) {
        v16bf a0 = load_frag(x1 + (size_t)row0 * 1024 + k0, 1024);
        v16bf a1 = load_frag(x1 + (size_t)(row0 + 16) * 1024 + k0, 1024);
#pragma unroll
        for (int j = 0; j < 4; ++j) {
            v16bf b = load_frag(W + (size_t)(n0 + j * 16) * 2048 + 1024 + k0, 2048);
            acc0[j] = wmma_bf16(a0, b, acc0[j]);
            acc1[j] = wmma_bf16(a1, b, acc1[j]);
        }
    }
    unsigned l  = lane_id();
    unsigned nn = l & 15u;
    unsigned mo = (l >> 4) * 8u;
#pragma unroll
    for (int j = 0; j < 4; ++j) {
        int n = n0 + j * 16 + nn;
        float bs = bias[n];
#pragma unroll
        for (int r = 0; r < 8; ++r) {
            float v0 = acc0[j][r] + bs;
            float v1 = acc1[j][r] + bs;
            if (relu_flag) { v0 = fmaxf(v0, 0.0f); v1 = fmaxf(v1, 0.0f); }
            out[(size_t)(row0 + mo + r) * N + n]      = (bf16)v0;
            out[(size_t)(row0 + 16 + mo + r) * N + n] = (bf16)v1;
        }
    }
}

// ---------------------------------------------------------------------------
// Step kernel 4: fused GRU cell. Each wave computes a 32x16 tile of h_new,
// chaining 12 WMMA accumulators (r,z,n gates x {W_ih@gru_in, W_hh@h} x 2 row
// tiles); the 6 weight fragments per k-tile are reused across both row tiles.
// ---------------------------------------------------------------------------
__global__ void __launch_bounds__(256)
k_gru_cell(const bf16* __restrict__ xg, const bf16* __restrict__ hb,
           const float* __restrict__ hf,
           const bf16* __restrict__ Wih, const bf16* __restrict__ Whh,
           const float* __restrict__ bih, const float* __restrict__ bhh,
           bf16* __restrict__ h_bf_new, float* __restrict__ h_f32_new) {
    const int row0 = blockIdx.x * 32;
    const int n0   = (blockIdx.y * 8 + (threadIdx.x >> 5)) * 16;

    v8f gir0 = {}, giz0 = {}, gin0 = {}, ghr0 = {}, ghz0 = {}, ghn0 = {};
    v8f gir1 = {}, giz1 = {}, gin1 = {}, ghr1 = {}, ghz1 = {}, ghn1 = {};
    for (int k0 = 0; k0 < 1024; k0 += 32) {
        v16bf ax0 = load_frag(xg + (size_t)row0 * 1024 + k0, 1024);
        v16bf ax1 = load_frag(xg + (size_t)(row0 + 16) * 1024 + k0, 1024);
        v16bf ah0 = load_frag(hb + (size_t)row0 * 1024 + k0, 1024);
        v16bf ah1 = load_frag(hb + (size_t)(row0 + 16) * 1024 + k0, 1024);
        v16bf br = load_frag(Wih + (size_t)(n0)        * 1024 + k0, 1024);
        v16bf bz = load_frag(Wih + (size_t)(1024 + n0) * 1024 + k0, 1024);
        v16bf bn = load_frag(Wih + (size_t)(2048 + n0) * 1024 + k0, 1024);
        gir0 = wmma_bf16(ax0, br, gir0);  gir1 = wmma_bf16(ax1, br, gir1);
        giz0 = wmma_bf16(ax0, bz, giz0);  giz1 = wmma_bf16(ax1, bz, giz1);
        gin0 = wmma_bf16(ax0, bn, gin0);  gin1 = wmma_bf16(ax1, bn, gin1);
        v16bf cr = load_frag(Whh + (size_t)(n0)        * 1024 + k0, 1024);
        v16bf cz = load_frag(Whh + (size_t)(1024 + n0) * 1024 + k0, 1024);
        v16bf cn = load_frag(Whh + (size_t)(2048 + n0) * 1024 + k0, 1024);
        ghr0 = wmma_bf16(ah0, cr, ghr0);  ghr1 = wmma_bf16(ah1, cr, ghr1);
        ghz0 = wmma_bf16(ah0, cz, ghz0);  ghz1 = wmma_bf16(ah1, cz, ghz1);
        ghn0 = wmma_bf16(ah0, cn, ghn0);  ghn1 = wmma_bf16(ah1, cn, ghn1);
    }
    unsigned l  = lane_id();
    unsigned nn = l & 15u;
    unsigned mo = (l >> 4) * 8u;
    int j = n0 + nn;
    float bir = bih[j], biz = bih[1024 + j], bin = bih[2048 + j];
    float bhr = bhh[j], bhz = bhh[1024 + j], bhn = bhh[2048 + j];
#pragma unroll
    for (int r = 0; r < 8; ++r) {
        {   // row tile 0
            int row = row0 + mo + r;
            float rg = sigmoidf_fast(gir0[r] + bir + ghr0[r] + bhr);
            float zg = sigmoidf_fast(giz0[r] + biz + ghz0[r] + bhz);
            float ng = tanhf(gin0[r] + bin + rg * (ghn0[r] + bhn));
            float ho = hf[(size_t)row * 1024 + j];
            float hn = (1.0f - zg) * ng + zg * ho;
            h_f32_new[(size_t)row * 1024 + j] = hn;
            h_bf_new[(size_t)row * 1024 + j]  = (bf16)hn;
        }
        {   // row tile 1
            int row = row0 + 16 + mo + r;
            float rg = sigmoidf_fast(gir1[r] + bir + ghr1[r] + bhr);
            float zg = sigmoidf_fast(giz1[r] + biz + ghz1[r] + bhz);
            float ng = tanhf(gin1[r] + bin + rg * (ghn1[r] + bhn));
            float ho = hf[(size_t)row * 1024 + j];
            float hn = (1.0f - zg) * ng + zg * ho;
            h_f32_new[(size_t)row * 1024 + j] = hn;
            h_bf_new[(size_t)row * 1024 + j]  = (bf16)hn;
        }
    }
}

// ---------------------------------------------------------------------------
// Step kernel 5: out_tok = h_new @ Wo^T + bo. Writes f32 straight into the
// [B,T,O] output slab (row stride T*O, non-temporal: pure HBM stream, keep L2
// for the resident encoder/weights) plus the bf16 dec_in for the next step.
// ---------------------------------------------------------------------------
__global__ void __launch_bounds__(256)
k_out_proj(const bf16* __restrict__ hb_new, const bf16* __restrict__ Wo,
           const float* __restrict__ bo, float* __restrict__ out_t,
           bf16* __restrict__ dec_bf) {
    const int row0 = blockIdx.x * 32;
    const int n0   = (blockIdx.y * 8 + (threadIdx.x >> 5)) * 64;
    v8f acc0[4] = {{}, {}, {}, {}};
    v8f acc1[4] = {{}, {}, {}, {}};

    for (int k0 = 0; k0 < 1024; k0 += 32) {
        v16bf a0 = load_frag(hb_new + (size_t)row0 * 1024 + k0, 1024);
        v16bf a1 = load_frag(hb_new + (size_t)(row0 + 16) * 1024 + k0, 1024);
#pragma unroll
        for (int j = 0; j < 4; ++j) {
            v16bf b = load_frag(Wo + (size_t)(n0 + j * 16) * 1024 + k0, 1024);
            acc0[j] = wmma_bf16(a0, b, acc0[j]);
            acc1[j] = wmma_bf16(a1, b, acc1[j]);
        }
    }
    unsigned l  = lane_id();
    unsigned nn = l & 15u;
    unsigned mo = (l >> 4) * 8u;
#pragma unroll
    for (int j = 0; j < 4; ++j) {
        int n = n0 + j * 16 + nn;
        float bs = bo[n];
#pragma unroll
        for (int r = 0; r < 8; ++r) {
            int rowa = row0 + mo + r;
            int rowb = row0 + 16 + mo + r;
            float v0 = acc0[j][r] + bs;
            float v1 = acc1[j][r] + bs;
            __builtin_nontemporal_store(v0, &out_t[(size_t)rowa * (TSTEP * 1024) + n]);
            __builtin_nontemporal_store(v1, &out_t[(size_t)rowb * (TSTEP * 1024) + n]);
            dec_bf[(size_t)rowa * 1024 + n] = (bf16)v0;
            dec_bf[(size_t)rowb * 1024 + n] = (bf16)v1;
        }
    }
}

// ---------------------------------------------------------------------------
// Host launcher
// ---------------------------------------------------------------------------
extern "C" void kernel_launch(void* const* d_in, const int* in_sizes, int n_in,
                              void* d_out, int out_size, void* d_ws, size_t ws_size,
                              hipStream_t stream) {
    (void)in_sizes; (void)n_in; (void)out_size; (void)ws_size;

    const float* enc    = (const float*)d_in[0];   // [B,S,A]
    const float* hidden = (const float*)d_in[1];   // [1,B,H]
    const float* Wa     = (const float*)d_in[3];   // [64, 2048]
    const float* ba     = (const float*)d_in[4];
    const float* Wc     = (const float*)d_in[5];   // [1024, 2048]
    const float* bc     = (const float*)d_in[6];
    const float* Wih    = (const float*)d_in[7];   // [3072, 1024]
    const float* Whh    = (const float*)d_in[8];
    const float* bih    = (const float*)d_in[9];
    const float* bhh    = (const float*)d_in[10];
    const float* Wo     = (const float*)d_in[11];  // [1024, 1024]
    const float* bo     = (const float*)d_in[12];
    float* out = (float*)d_out;

    // --- carve workspace ---------------------------------------------------
    uint8_t* base = (uint8_t*)d_ws;
    size_t off = 0;
    auto carve = [&](size_t bytes) -> uint8_t* {
        uint8_t* p = base + off;
        off = (off + bytes + 255) & ~(size_t)255;
        return p;
    };
    bf16* Wa_bf  = (bf16*)carve((size_t)64 * 2048 * 2);
    bf16* Wc_bf  = (bf16*)carve((size_t)1024 * 2048 * 2);
    bf16* Wih_bf = (bf16*)carve((size_t)3072 * 1024 * 2);
    bf16* Whh_bf = (bf16*)carve((size_t)3072 * 1024 * 2);
    bf16* Wo_bf  = (bf16*)carve((size_t)1024 * 1024 * 2);
    bf16* dec_bf = (bf16*)carve((size_t)BATCH * 1024 * 2);
    bf16* attn_bf= (bf16*)carve((size_t)BATCH * 1024 * 2);
    bf16* gru_bf = (bf16*)carve((size_t)BATCH * 1024 * 2);
    bf16* h_bf[2]; float* h_f[2];
    h_bf[0] = (bf16*)carve((size_t)BATCH * 1024 * 2);
    h_bf[1] = (bf16*)carve((size_t)BATCH * 1024 * 2);
    h_f[0]  = (float*)carve((size_t)BATCH * 1024 * 4);
    h_f[1]  = (float*)carve((size_t)BATCH * 1024 * 4);
    float* attn_w = (float*)carve((size_t)BATCH * 64 * 4);

    auto cvt = [&](const float* s, bf16* d, int n) {
        k_f32_to_bf16<<<(n + 255) / 256, 256, 0, stream>>>(s, d, n);
    };

    // --- one-time weight conversion + state init ---------------------------
    cvt(Wa, Wa_bf, 64 * 2048);
    cvt(Wc, Wc_bf, 1024 * 2048);
    cvt(Wih, Wih_bf, 3072 * 1024);
    cvt(Whh, Whh_bf, 3072 * 1024);
    cvt(Wo, Wo_bf, 1024 * 1024);
    cvt(hidden, h_bf[0], BATCH * 1024);
    k_zero_bf16<<<(BATCH * 1024 + 255) / 256, 256, 0, stream>>>(dec_bf, BATCH * 1024);
    hipMemcpyAsync(h_f[0], hidden, (size_t)BATCH * 1024 * 4,
                   hipMemcpyDeviceToDevice, stream);

    // --- 128 sequential decode steps ---------------------------------------
    for (int t = 0; t < TSTEP; ++t) {
        int cur = t & 1, nxt = cur ^ 1;
        k_attn_logits<<<dim3(BATCH / 16), 128, 0, stream>>>(
            dec_bf, h_bf[cur], Wa_bf, ba, attn_w);
        k_attn_apply<<<dim3(BATCH, 4), 256, 0, stream>>>(attn_w, enc, attn_bf);
        k_gemm_2seg<<<dim3(BATCH / 32, 2), 256, 0, stream>>>(
            dec_bf, attn_bf, Wc_bf, bc, gru_bf, 1024, /*relu=*/1);
        k_gru_cell<<<dim3(BATCH / 32, 8), 256, 0, stream>>>(
            gru_bf, h_bf[cur], h_f[cur], Wih_bf, Whh_bf, bih, bhh,
            h_bf[nxt], h_f[nxt]);
        k_out_proj<<<dim3(BATCH / 32, 2), 256, 0, stream>>>(
            h_bf[nxt], Wo_bf, bo, out + (size_t)t * 1024, dec_bf);
    }

    // --- final hidden state: tuple output tail [1,B,H] ----------------------
    hipMemcpyAsync(out + (size_t)BATCH * TSTEP * 1024, h_f[0],
                   (size_t)BATCH * 1024 * 4, hipMemcpyDeviceToDevice, stream);
}
